// Atom_Intra_Layer_62672162783754
// MI455X (gfx1250) — compile-verified, matrix operand order checked by reading, not run on hardware
//
#include <hip/hip_runtime.h>

typedef __attribute__((ext_vector_type(16))) _Float16 v16h;
typedef __attribute__((ext_vector_type(8)))  float    v8f;

#define FEAT_K   288        // 2*D + ED
#define DIM      128        // D and H*C
#define HEADS    8
#define CPH      16
#define MTILE    32         // edges per block in edge kernel (2 WMMA M-tiles)
#define FEAT_STRIDE 290     // halfs; 580B = 145 dwords (odd) -> conflict-free
#define HID_STRIDE  130     // halfs; 260B = 65 dwords (odd)
#define KOUT_STRIDE 132     // floats

__device__ __forceinline__ v8f wmma_f16(v16h a, v16h b, v8f c) {
    // V_WMMA_F32_16X16X32_F16 : D = A(16x32 f16) x B(32x16 f16) + C(16x16 f32)
    return __builtin_amdgcn_wmma_f32_16x16x32_f16(
        false, a, false, b, (short)0, c, false, false);
}

// A-fragment (16x32, f16) from row-major LDS tile: lane l holds row m=l&15.
// Half-pair p maps to K = (p&3)*2 + (p>>2)*16 + (l>>4)*8  (per ISA 7.12.2)
__device__ __forceinline__ v16h load_frag_row(const _Float16* base, int strideHalfs, int k0) {
    const int l  = threadIdx.x & 31;
    const int m  = l & 15;
    const int hi = l >> 4;
    union { v16h v; unsigned u[8]; } f;
    const _Float16* row = base + m * strideHalfs + k0 + hi * 8;
#pragma unroll
    for (int p = 0; p < 8; ++p) {
        const int k = ((p & 3) * 2) + ((p >> 2) * 16);
        f.u[p] = *(const unsigned*)(row + k);
    }
    return f.v;
}

// B-fragment (32x16, f16) from transposed weights WT[N][K] (row-major, row = output col n).
__device__ __forceinline__ v16h load_frag_col(const _Float16* __restrict__ wT, int strideHalfs,
                                              int n0, int k0) {
    const int l  = threadIdx.x & 31;
    const int n  = n0 + (l & 15);
    const int hi = l >> 4;
    union { v16h v; unsigned u[8]; } f;
    const _Float16* row = wT + n * strideHalfs + k0 + hi * 8;
#pragma unroll
    for (int p = 0; p < 8; ++p) {
        const int k = ((p & 3) * 2) + ((p >> 2) * 16);
        f.u[p] = *(const unsigned*)(row + k);
    }
    return f.v;
}

// Order-preserving uint encoding of float for atomic max
__device__ __forceinline__ unsigned encf(float f) {
    unsigned u = __float_as_uint(f);
    return (u & 0x80000000u) ? ~u : (u | 0x80000000u);
}
__device__ __forceinline__ float decf(unsigned u) {
    return __uint_as_float((u & 0x80000000u) ? (u ^ 0x80000000u) : ~u);
}

__device__ __forceinline__ float silu(float x) {
    return x / (1.0f + __expf(-x));
}

// ---------------------------------------------------------------------------
// Kernel 1: transpose+convert all 5 weight matrices to f16 [N][K]
__global__ __launch_bounds__(256)
void convert_weights(const float* __restrict__ Wq,  const float* __restrict__ Wk1,
                     const float* __restrict__ Wk2, const float* __restrict__ Wv1,
                     const float* __restrict__ Wv2,
                     _Float16* __restrict__ WqT,  _Float16* __restrict__ Wk1T,
                     _Float16* __restrict__ Wk2T, _Float16* __restrict__ Wv1T,
                     _Float16* __restrict__ Wv2T) {
    const int idx = blockIdx.x * blockDim.x + threadIdx.x;
    if (idx < DIM * FEAT_K) {                 // 128 x 288
        const int n = idx / FEAT_K, k = idx % FEAT_K;
        Wk1T[idx] = (_Float16)Wk1[k * DIM + n];
        Wv1T[idx] = (_Float16)Wv1[k * DIM + n];
    }
    if (idx < DIM * DIM) {                    // 128 x 128
        const int n = idx / DIM, k = idx % DIM;
        WqT[idx]  = (_Float16)Wq [k * DIM + n];
        Wk2T[idx] = (_Float16)Wk2[k * DIM + n];
        Wv2T[idx] = (_Float16)Wv2[k * DIM + n];
    }
}

// Kernel 2: init output, seg_max (-inf encoded), seg_den (0)
__global__ __launch_bounds__(256)
void init_kernel(float* __restrict__ out, unsigned* __restrict__ seg_max,
                 float* __restrict__ seg_den, int N) {
    const int idx = blockIdx.x * blockDim.x + threadIdx.x;
    if (idx < N * DIM) out[idx] = 0.0f;
    if (idx < N * HEADS) { seg_max[idx] = 0x007FFFFFu; /* enc(-inf) */ seg_den[idx] = 0.0f; }
}

// Kernel 3: query = x @ Wq + bq   (16 rows / block, 8 waves = 8 col-tiles)
__global__ __launch_bounds__(256)
void query_kernel(const float* __restrict__ x, const _Float16* __restrict__ WqT,
                  const float* __restrict__ bq, float* __restrict__ q, int N) {
    __shared__ _Float16 lds_x[16 * HID_STRIDE];
    const int t  = threadIdx.x;
    const int n0 = blockIdx.x * 16;
    const bool full = (n0 + 16 <= N);
#pragma unroll
    for (int i = 0; i < 8; ++i) {
        const int idx = t + 256 * i;                 // 2048 elements
        const int r = idx >> 7, c = idx & 127;
        float v = (full || n0 + r < N) ? x[(size_t)(n0 + r) * DIM + c] : 0.0f;
        lds_x[r * HID_STRIDE + c] = (_Float16)v;
    }
    __syncthreads();

    const int w = t >> 5;
    v8f acc = {};
#pragma unroll
    for (int kk = 0; kk < 4; ++kk) {
        v16h a = load_frag_row(lds_x, HID_STRIDE, kk * 32);
        v16h b = load_frag_col(WqT, DIM, w * 16, kk * 32);
        acc = wmma_f16(a, b, acc);
    }
    const int l = t & 31, c = l & 15, hi = l >> 4;
    const int n = w * 16 + c;
    const float bb = bq[n];
    if (full) {
#pragma unroll
        for (int r = 0; r < 8; ++r)
            q[(size_t)(n0 + r + 8 * hi) * DIM + n] = acc[r] + bb;
    } else {
#pragma unroll
        for (int r = 0; r < 8; ++r) {
            const int m = r + 8 * hi;
            if (n0 + m < N) q[(size_t)(n0 + m) * DIM + n] = acc[r] + bb;
        }
    }
}

// Kernel 4: main per-edge kernel (gather + LN + 2x 2-layer MLP + logits)
// 32 edges / block: 2 WMMA M-tiles per wave => every B fragment feeds 2 WMMAs.
__global__ __launch_bounds__(256)
void edge_main_kernel(const float* __restrict__ x, const int* __restrict__ ei,
                      const float* __restrict__ eattr,
                      const float* __restrict__ ln_g, const float* __restrict__ ln_b,
                      const _Float16* __restrict__ Wk1T, const float* __restrict__ bk1,
                      const _Float16* __restrict__ Wk2T, const float* __restrict__ bk2,
                      const _Float16* __restrict__ Wv1T, const float* __restrict__ bv1,
                      const _Float16* __restrict__ Wv2T, const float* __restrict__ bv2,
                      const float* __restrict__ q,
                      _Float16* __restrict__ val_out,   // [E][128] f16
                      float* __restrict__ logits,       // [E][8]
                      unsigned* __restrict__ seg_max,   // [N][8] encoded
                      int E) {
    __shared__ _Float16 lds_feat[MTILE * FEAT_STRIDE];
    __shared__ _Float16 lds_hk[MTILE * HID_STRIDE];
    __shared__ _Float16 lds_hv[MTILE * HID_STRIDE];
    __shared__ float    lds_kout[MTILE * KOUT_STRIDE];
    __shared__ int      lds_src[MTILE];

    const int t  = threadIdx.x;
    const int e0 = blockIdx.x * MTILE;
    const bool full = (e0 + MTILE <= E);

    // ---- Phase A: gather [eattr | x[i0] | x[i1]] and LayerNorm -> f16 LDS
    {
        const int mg = t >> 4;        // 16 groups; each handles edges mg, mg+16
        const int j  = t & 15;
#pragma unroll
        for (int halfm = 0; halfm < 2; ++halfm) {
            const int m = mg + halfm * 16;
            const int e = full ? (e0 + m) : min(e0 + m, E - 1);
            const int i0 = ei[e];
            const int i1 = ei[E + e];
            if (j == 0) lds_src[m] = i0;
            float r[18];
            float s = 0.0f, s2 = 0.0f;
#pragma unroll
            for (int k = 0; k < 18; ++k) {
                const int f = j + 16 * k;  // coalesced across the 16-lane group
                float v;
                if (f < 32)       v = eattr[(size_t)e * 32 + f];
                else if (f < 160) v = x[(size_t)i0 * DIM + (f - 32)];
                else              v = x[(size_t)i1 * DIM + (f - 160)];
                r[k] = v; s += v; s2 += v * v;
            }
#pragma unroll
            for (int ofs = 8; ofs >= 1; ofs >>= 1) {
                s  += __shfl_xor(s,  ofs, 16);
                s2 += __shfl_xor(s2, ofs, 16);
            }
            const float mean = s * (1.0f / FEAT_K);
            const float var  = s2 * (1.0f / FEAT_K) - mean * mean;
            const float rstd = rsqrtf(var + 1e-5f);
#pragma unroll
            for (int k = 0; k < 18; ++k) {
                const int f = j + 16 * k;
                const float nv = (r[k] - mean) * rstd * ln_g[f] + ln_b[f];
                lds_feat[m * FEAT_STRIDE + f] = (_Float16)nv;
            }
        }
    }
    __syncthreads();

    const int w  = t >> 5;          // wave id = output col tile
    const int l  = t & 31;
    const int c  = l & 15;
    const int hi = l >> 4;
    const int n  = w * 16 + c;

    // ---- Phase B: GEMM1 (288 -> 128), key & value, 2 M-tiles, + bias + SiLU
    {
        v8f ck0 = {}, ck1 = {}, cv0 = {}, cv1 = {};
#pragma unroll
        for (int kk = 0; kk < 9; ++kk) {
            const int k0 = kk * 32;
            v16h a0 = load_frag_row(lds_feat, FEAT_STRIDE, k0);
            v16h a1 = load_frag_row(lds_feat + 16 * FEAT_STRIDE, FEAT_STRIDE, k0);
            v16h bk = load_frag_col(Wk1T, FEAT_K, w * 16, k0);
            ck0 = wmma_f16(a0, bk, ck0);
            ck1 = wmma_f16(a1, bk, ck1);
            v16h bv = load_frag_col(Wv1T, FEAT_K, w * 16, k0);
            cv0 = wmma_f16(a0, bv, cv0);
            cv1 = wmma_f16(a1, bv, cv1);
        }
        const float bbk = bk1[n], bbv = bv1[n];
#pragma unroll
        for (int r = 0; r < 8; ++r) {
            const int m = r + 8 * hi;
            lds_hk[m * HID_STRIDE + n]          = (_Float16)silu(ck0[r] + bbk);
            lds_hk[(m + 16) * HID_STRIDE + n]   = (_Float16)silu(ck1[r] + bbk);
            lds_hv[m * HID_STRIDE + n]          = (_Float16)silu(cv0[r] + bbv);
            lds_hv[(m + 16) * HID_STRIDE + n]   = (_Float16)silu(cv1[r] + bbv);
        }
    }
    __syncthreads();

    // ---- Phase C: GEMM2 (128 -> 128), key & value, 2 M-tiles, + bias
    {
        v8f dk0 = {}, dk1 = {}, dv0 = {}, dv1 = {};
#pragma unroll
        for (int kk = 0; kk < 4; ++kk) {
            const int k0 = kk * 32;
            v16h ak0 = load_frag_row(lds_hk, HID_STRIDE, k0);
            v16h ak1 = load_frag_row(lds_hk + 16 * HID_STRIDE, HID_STRIDE, k0);
            v16h bk  = load_frag_col(Wk2T, DIM, w * 16, k0);
            dk0 = wmma_f16(ak0, bk, dk0);
            dk1 = wmma_f16(ak1, bk, dk1);
            v16h av0 = load_frag_row(lds_hv, HID_STRIDE, k0);
            v16h av1 = load_frag_row(lds_hv + 16 * HID_STRIDE, HID_STRIDE, k0);
            v16h bv  = load_frag_col(Wv2T, DIM, w * 16, k0);
            dv0 = wmma_f16(av0, bv, dv0);
            dv1 = wmma_f16(av1, bv, dv1);
        }
        const float bbk = bk2[n], bbv = bv2[n];
        if (full) {
#pragma unroll
            for (int r = 0; r < 8; ++r) {
                const int m = r + 8 * hi;
                lds_kout[m * KOUT_STRIDE + n]        = dk0[r] + bbk;
                lds_kout[(m + 16) * KOUT_STRIDE + n] = dk1[r] + bbk;
                val_out[(size_t)(e0 + m) * DIM + n]      = (_Float16)(dv0[r] + bbv);
                val_out[(size_t)(e0 + m + 16) * DIM + n] = (_Float16)(dv1[r] + bbv);
            }
        } else {
#pragma unroll
            for (int r = 0; r < 8; ++r) {
                const int m = r + 8 * hi;
                lds_kout[m * KOUT_STRIDE + n]        = dk0[r] + bbk;
                lds_kout[(m + 16) * KOUT_STRIDE + n] = dk1[r] + bbk;
                if (e0 + m < E)
                    val_out[(size_t)(e0 + m) * DIM + n] = (_Float16)(dv0[r] + bbv);
                if (e0 + m + 16 < E)
                    val_out[(size_t)(e0 + m + 16) * DIM + n] = (_Float16)(dv1[r] + bbv);
            }
        }
    }
    __syncthreads();

    // ---- Phase D: logits[m][h] = (q[src] . k) / sqrt(C); atomic seg-max
    {
        const int m = t >> 3, h = t & 7;     // 32 edges x 8 heads = 256 threads
        if (full || e0 + m < E) {
            const int src = lds_src[m];
            float acc = 0.0f;
#pragma unroll
            for (int cc = 0; cc < CPH; ++cc)
                acc += q[(size_t)src * DIM + h * CPH + cc] *
                       lds_kout[m * KOUT_STRIDE + h * CPH + cc];
            const float lg = acc * 0.25f;               // 1/sqrt(16)
            logits[(size_t)(e0 + m) * HEADS + h] = lg;
            atomicMax(&seg_max[(size_t)src * HEADS + h], encf(lg));
        }
    }
}

// Kernel 5: ex = exp(logit - segmax); denom += ex
__global__ __launch_bounds__(256)
void exp_kernel(const int* __restrict__ ei, float* __restrict__ logits,
                const unsigned* __restrict__ seg_max, float* __restrict__ seg_den, int E) {
    const int idx = blockIdx.x * blockDim.x + threadIdx.x;
    if (idx >= E * HEADS) return;
    const int e = idx >> 3, h = idx & 7;
    const int src = ei[e];
    const float mx = decf(seg_max[(size_t)src * HEADS + h]);
    const float ex = __expf(logits[idx] - mx);
    logits[idx] = ex;
    atomicAdd(&seg_den[(size_t)src * HEADS + h], ex);
}

// Kernel 6: out[src] += (ex/denom) * v
__global__ __launch_bounds__(256)
void scatter_kernel(const int* __restrict__ ei, const float* __restrict__ logits,
                    const float* __restrict__ seg_den, const _Float16* __restrict__ val,
                    float* __restrict__ out, int E) {
    const size_t idx = (size_t)blockIdx.x * blockDim.x + threadIdx.x;
    if (idx >= (size_t)E * DIM) return;
    const int e = (int)(idx >> 7);
    const int c = (int)(idx & 127);
    const int h = c >> 4;
    const int src = ei[e];
    const float ex  = logits[(size_t)e * HEADS + h];
    const float den = seg_den[(size_t)src * HEADS + h];
    const float alpha = ex / (den + 1e-16f);
    atomicAdd(&out[(size_t)src * DIM + c], alpha * (float)val[idx]);
}

// ---------------------------------------------------------------------------
extern "C" void kernel_launch(void* const* d_in, const int* in_sizes, int n_in,
                              void* d_out, int out_size, void* d_ws, size_t ws_size,
                              hipStream_t stream) {
    const float* x     = (const float*)d_in[0];
    const int*   ei    = (const int*)  d_in[1];   // [2][E]
    const float* eattr = (const float*)d_in[2];
    const float* ln_g  = (const float*)d_in[3];
    const float* ln_b  = (const float*)d_in[4];
    const float* Wq    = (const float*)d_in[5];
    const float* bq    = (const float*)d_in[6];
    const float* Wk1   = (const float*)d_in[7];
    const float* bk1   = (const float*)d_in[8];
    const float* Wk2   = (const float*)d_in[9];
    const float* bk2   = (const float*)d_in[10];
    const float* Wv1   = (const float*)d_in[11];
    const float* bv1   = (const float*)d_in[12];
    const float* Wv2   = (const float*)d_in[13];
    const float* bv2   = (const float*)d_in[14];
    float* out = (float*)d_out;

    const int N = in_sizes[0] / DIM;
    const int E = in_sizes[1] / 2;

    // Workspace carve-up (256B aligned)
    char* ws = (char*)d_ws;
    size_t off = 0;
    auto carve = [&](size_t bytes) { char* p = ws + off; off = (off + bytes + 255) & ~(size_t)255; return p; };
    _Float16* WqT   = (_Float16*)carve((size_t)DIM * DIM    * 2);
    _Float16* Wk1T  = (_Float16*)carve((size_t)DIM * FEAT_K * 2);
    _Float16* Wk2T  = (_Float16*)carve((size_t)DIM * DIM    * 2);
    _Float16* Wv1T  = (_Float16*)carve((size_t)DIM * FEAT_K * 2);
    _Float16* Wv2T  = (_Float16*)carve((size_t)DIM * DIM    * 2);
    float*    qbuf  = (float*)   carve((size_t)N * DIM * 4);
    _Float16* valb  = (_Float16*)carve((size_t)E * DIM * 2);
    float*    lgb   = (float*)   carve((size_t)E * HEADS * 4);
    unsigned* smax  = (unsigned*)carve((size_t)N * HEADS * 4);
    float*    sden  = (float*)   carve((size_t)N * HEADS * 4);

    convert_weights<<<(DIM * FEAT_K + 255) / 256, 256, 0, stream>>>(
        Wq, Wk1, Wk2, Wv1, Wv2, WqT, Wk1T, Wk2T, Wv1T, Wv2T);

    init_kernel<<<(N * DIM + 255) / 256, 256, 0, stream>>>(out, smax, sden, N);

    query_kernel<<<(N + 15) / 16, 256, 0, stream>>>(x, WqT, bq, qbuf, N);

    edge_main_kernel<<<(E + MTILE - 1) / MTILE, 256, 0, stream>>>(
        x, ei, eattr, ln_g, ln_b,
        Wk1T, bk1, Wk2T, bk2, Wv1T, bv1, Wv2T, bv2,
        qbuf, valb, lgb, smax, E);

    exp_kernel<<<(E * HEADS + 255) / 256, 256, 0, stream>>>(ei, lgb, smax, sden, E);

    scatter_kernel<<<(int)(((size_t)E * DIM + 255) / 256), 256, 0, stream>>>(
        ei, lgb, sden, valb, out, E);
}
